// Mamba2_6949257084997
// MI455X (gfx1250) — compile-verified, matrix operand order checked by reading, not run on hardware
//
#include <hip/hip_runtime.h>
#include <hip/hip_bf16.h>
#include <stdint.h>
#include <stddef.h>

// ---- model constants -------------------------------------------------------
#define D_MODEL   768
#define D_INNER   1536
#define D_STATE   128
#define D_CONV    4
#define HEADDIM   64
#define NHEADS    24
#define CONV_DIM  (D_INNER + 2*D_STATE)              // 1792
#define D_IN_PROJ (2*D_INNER + 2*D_STATE + NHEADS)   // 3352
#define EPS_RMS   1e-5f
#define BATCH     4
#define SEQLEN    2048
#define NTOK      (BATCH*SEQLEN)                     // 8192

typedef __attribute__((ext_vector_type(16))) __bf16 v16bf;
typedef __attribute__((ext_vector_type(8)))  float  v8f;

// round-to-nearest-even fp32 -> bf16 (bit ops; no __bf16 scalar arithmetic)
__device__ __forceinline__ unsigned short f32_to_bf16(float f) {
  unsigned int x = __float_as_uint(f);
  x += 0x7fffu + ((x >> 16) & 1u);
  return (unsigned short)(x >> 16);
}

// ---- fp32 -> bf16 bulk convert (n4 = number of float4 groups) --------------
__global__ __launch_bounds__(256) void cvt_f32_bf16(const float* __restrict__ src,
                                                    unsigned short* __restrict__ dst,
                                                    int n4) {
  int i = blockIdx.x * 256 + threadIdx.x;
  if (i >= n4) return;
  float4 v = ((const float4*)src)[i];
  unsigned int lo = (unsigned int)f32_to_bf16(v.x) | ((unsigned int)f32_to_bf16(v.y) << 16);
  unsigned int hi = (unsigned int)f32_to_bf16(v.z) | ((unsigned int)f32_to_bf16(v.w) << 16);
  ((uint2*)dst)[i] = make_uint2(lo, hi);
}

// ---- bf16 WMMA GEMM: C[M][N] = A[M][K] * B[N][K]^T -------------------------
// block = 256 threads (8 waves), tile 128(M) x 64(N), K-step 32.
#define LSTR 40   // LDS row stride in bf16 units (32 data + 8 pad, keeps 16B align)

__device__ __forceinline__ v16bf load_frag(const unsigned short* p, int lane) {
  // CDNA5 16-bit A/B fragment: lane<16 -> row=lane, K 0..7 & 16..23;
  // lane>=16 -> row=lane-16, +8 K offset.  Two 16B LDS loads per lane.
  int half = lane >> 4;
  int r    = lane & 15;
  const uint4* q = (const uint4*)(p + r * LSTR + half * 8);
  union { v16bf v; uint4 u[2]; } frag;
  frag.u[0] = q[0];        // K (0..7)+8*half
  frag.u[1] = q[2];        // +32 bytes: K (16..23)+8*half
  return frag.v;
}

__global__ __launch_bounds__(256) void gemm_bf16(const unsigned short* __restrict__ A,
                                                 const unsigned short* __restrict__ B,
                                                 float* __restrict__ C,
                                                 int M, int N, int K) {
  __shared__ unsigned short sA[128 * LSTR];
  __shared__ unsigned short sB[64 * LSTR];
  const int tid  = threadIdx.x;
  const int lane = tid & 31;
  const int wave = tid >> 5;
  const int wm = wave & 3;        // 4 waves along M (32 rows each)
  const int wn = wave >> 2;       // 2 waves along N (32 cols each)
  const int row0 = blockIdx.x * 128;
  const int col0 = blockIdx.y * 64;

  v8f acc00 = {}, acc01 = {}, acc10 = {}, acc11 = {};

  const int ar = tid >> 1;          // 0..127: A row within tile
  const int ah = (tid & 1) * 16;    // which 16-bf16 half of the K-step

  for (int k0 = 0; k0 < K; k0 += 32) {
    { // stage A tile (128x32 bf16), 2x global_load_b128 per thread
      const uint4* g = (const uint4*)(A + (size_t)(row0 + ar) * K + k0 + ah);
      uint4 d0 = g[0], d1 = g[1];
      uint4* s = (uint4*)(sA + ar * LSTR + ah);
      s[0] = d0; s[1] = d1;
    }
    if (tid < 128) { // stage B tile (64x32 bf16), zero-fill past N
      int r = tid >> 1;
      int h = (tid & 1) * 16;
      int gr = col0 + r;
      uint4 d0 = {}, d1 = {};
      if (gr < N) {
        const uint4* g = (const uint4*)(B + (size_t)gr * K + k0 + h);
        d0 = g[0]; d1 = g[1];
      }
      uint4* s = (uint4*)(sB + r * LSTR + h);
      s[0] = d0; s[1] = d1;
    }
    if (k0 + 32 < K)  // pull next A K-tile toward the WGP (global_prefetch_b8)
      __builtin_prefetch(A + (size_t)(row0 + ar) * K + (k0 + 32) + ah, 0, 3);
    __syncthreads();

    const unsigned short* pa = sA + (wm * 32) * LSTR;
    const unsigned short* pb = sB + (wn * 32) * LSTR;
    v16bf a0 = load_frag(pa, lane);
    v16bf a1 = load_frag(pa + 16 * LSTR, lane);
    v16bf b0 = load_frag(pb, lane);
    v16bf b1 = load_frag(pb + 16 * LSTR, lane);
    acc00 = __builtin_amdgcn_wmma_f32_16x16x32_bf16(false, a0, false, b0, (short)0, acc00, false, false);
    acc01 = __builtin_amdgcn_wmma_f32_16x16x32_bf16(false, a0, false, b1, (short)0, acc01, false, false);
    acc10 = __builtin_amdgcn_wmma_f32_16x16x32_bf16(false, a1, false, b0, (short)0, acc10, false, false);
    acc11 = __builtin_amdgcn_wmma_f32_16x16x32_bf16(false, a1, false, b1, (short)0, acc11, false, false);
    __syncthreads();
  }

  // store: VGPR v of 16x16 f32 tile -> rows (v + 8*half), cols = lane&15
  const int half = lane >> 4;
  const int ln   = lane & 15;
  v8f accs[4] = {acc00, acc01, acc10, acc11};
  #pragma unroll
  for (int i = 0; i < 2; ++i) {
    #pragma unroll
    for (int j = 0; j < 2; ++j) {
      v8f a = accs[i * 2 + j];
      int col = col0 + wn * 32 + j * 16 + ln;
      if (col < N) {
        int rbase = row0 + wm * 32 + i * 16 + half * 8;
        #pragma unroll
        for (int v = 0; v < 8; ++v)
          C[(size_t)(rbase + v) * N + col] = a[v];
      }
    }
  }
}

// ---- depthwise conv (window 4) over seq + bias + SiLU ----------------------
__global__ __launch_bounds__(256) void conv_silu(const float* __restrict__ zxbcdt,
                                                 const float* __restrict__ conv_w,
                                                 const float* __restrict__ conv_b,
                                                 float* __restrict__ xBC) {
  int idx = blockIdx.x * 256 + threadIdx.x;
  if (idx >= NTOK * CONV_DIM) return;
  int c  = idx % CONV_DIM;
  int bl = idx / CONV_DIM;
  int l  = bl % SEQLEN;
  const float* base = zxbcdt + (size_t)bl * D_IN_PROJ + D_INNER + c;
  float acc = conv_b[c];
  #pragma unroll
  for (int k = 0; k < D_CONV; ++k) {
    int lk = l + k - (D_CONV - 1);
    if (lk >= 0)
      acc = fmaf(base[(ptrdiff_t)(k - (D_CONV - 1)) * D_IN_PROJ], conv_w[c * D_CONV + k], acc);
  }
  float sig = 1.f / (1.f + __expf(-acc));
  xBC[idx] = acc * sig;
}

// ---- dt = softplus(dt_raw + bias); dA = exp(dt * -exp(A_log)) --------------
__global__ __launch_bounds__(256) void dt_dA_kernel(const float* __restrict__ zxbcdt,
                                                    const float* __restrict__ dt_bias,
                                                    const float* __restrict__ A_log,
                                                    float* __restrict__ dt_s,
                                                    float* __restrict__ dA_s) {
  int idx = blockIdx.x * 256 + threadIdx.x;
  if (idx >= NTOK * NHEADS) return;
  int hh = idx % NHEADS;
  int bl = idx / NHEADS;
  float v = zxbcdt[(size_t)bl * D_IN_PROJ + D_INNER + CONV_DIM + hh] + dt_bias[hh];
  float sp = (v > 20.f) ? v : __logf(1.f + __expf(v));
  dt_s[idx] = sp;
  dA_s[idx] = __expf(-sp * __expf(A_log[hh]));
}

// ---- selective scan: one workgroup per (batch, head) -----------------------
// state h[64][128] fp32 lives in registers: thread t owns p = t>>2, n = (t&3)*32 .. +31
__global__ __launch_bounds__(256) void scan_kernel(const float* __restrict__ xBC,
                                                   const float* __restrict__ dt_s,
                                                   const float* __restrict__ dA_s,
                                                   const float* __restrict__ Dp,
                                                   float* __restrict__ y) {
  const int bh = blockIdx.x;
  const int b  = bh / NHEADS;
  const int h  = bh % NHEADS;
  const int tid = threadIdx.x;
  const int p   = tid >> 2;
  const int nb  = (tid & 3) * 32;
  __shared__ float sBC[256];   // [0,128)=B_t, [128,256)=C_t
  __shared__ float sX[64];
  float hs[32];
  #pragma unroll
  for (int j = 0; j < 32; ++j) hs[j] = 0.f;
  const float Dh = Dp[h];

  for (int t = 0; t < SEQLEN; ++t) {
    const size_t row = (size_t)b * SEQLEN + t;
    const float* xr = xBC + row * CONV_DIM;
    sBC[tid] = xr[D_INNER + tid];
    if (tid < HEADDIM) sX[tid] = xr[h * HEADDIM + tid];
    __syncthreads();

    const float dA  = dA_s[row * NHEADS + h];
    const float dtv = dt_s[row * NHEADS + h];
    const float x   = sX[p];
    const float xdt = x * dtv;
    float yp = 0.f;
    #pragma unroll
    for (int j = 0; j < 32; j += 4) {
      float4 Bv = *(const float4*)(&sBC[nb + j]);
      float4 Cv = *(const float4*)(&sBC[128 + nb + j]);
      hs[j+0] = fmaf(hs[j+0], dA, xdt * Bv.x); yp = fmaf(hs[j+0], Cv.x, yp);
      hs[j+1] = fmaf(hs[j+1], dA, xdt * Bv.y); yp = fmaf(hs[j+1], Cv.y, yp);
      hs[j+2] = fmaf(hs[j+2], dA, xdt * Bv.z); yp = fmaf(hs[j+2], Cv.z, yp);
      hs[j+3] = fmaf(hs[j+3], dA, xdt * Bv.w); yp = fmaf(hs[j+3], Cv.w, yp);
    }
    // reduce the 4 n-blocks (lanes differing in bits 0..1 of wave32)
    yp += __shfl_xor(yp, 1);
    yp += __shfl_xor(yp, 2);
    if ((tid & 3) == 0)
      y[row * D_INNER + h * HEADDIM + p] = yp + Dh * x;
    __syncthreads();
  }
}

// ---- RMSNorm over D_INNER, gate by SiLU(z), emit bf16 for out_proj ---------
__global__ __launch_bounds__(256) void rmsnorm_gate(const float* __restrict__ y,
                                                    const float* __restrict__ zxbcdt,
                                                    const float* __restrict__ norm_w,
                                                    unsigned short* __restrict__ g_bf) {
  const int row = blockIdx.x;
  const int tid = threadIdx.x;
  __shared__ float red[256];
  const float* yr = y + (size_t)row * D_INNER;
  const float* zr = zxbcdt + (size_t)row * D_IN_PROJ;   // z = first D_INNER cols
  float v[6];
  float ss = 0.f;
  #pragma unroll
  for (int i = 0; i < 6; ++i) { v[i] = yr[tid + i * 256]; ss = fmaf(v[i], v[i], ss); }
  red[tid] = ss;
  __syncthreads();
  #pragma unroll
  for (int s = 128; s > 0; s >>= 1) {
    if (tid < s) red[tid] += red[tid + s];
    __syncthreads();
  }
  const float inv = rsqrtf(red[0] * (1.0f / D_INNER) + EPS_RMS);
  #pragma unroll
  for (int i = 0; i < 6; ++i) {
    int c = tid + i * 256;
    float z = zr[c];
    float sig = 1.f / (1.f + __expf(-z));
    g_bf[(size_t)row * D_INNER + c] = f32_to_bf16(norm_w[c] * v[i] * inv * (z * sig));
  }
}

// ---- launcher --------------------------------------------------------------
extern "C" void kernel_launch(void* const* d_in, const int* in_sizes, int n_in,
                              void* d_out, int out_size, void* d_ws, size_t ws_size,
                              hipStream_t stream) {
  const float* u       = (const float*)d_in[0];
  const float* W_in    = (const float*)d_in[1];
  const float* conv_w  = (const float*)d_in[2];
  const float* conv_b  = (const float*)d_in[3];
  const float* dt_bias = (const float*)d_in[4];
  const float* A_log   = (const float*)d_in[5];
  const float* Dparam  = (const float*)d_in[6];
  const float* norm_w  = (const float*)d_in[7];
  const float* W_out   = (const float*)d_in[8];
  float* out = (float*)d_out;
  (void)in_sizes; (void)n_in; (void)out_size; (void)ws_size;

  char* ws = (char*)d_ws;
  size_t off = 0;
  auto wsalloc = [&](size_t bytes) -> char* {
    char* p = ws + off;
    off = (off + bytes + 255) & ~(size_t)255;
    return p;
  };
  float*          zxbcdt  = (float*)         wsalloc((size_t)NTOK * D_IN_PROJ * 4);
  unsigned short* u_bf    = (unsigned short*)wsalloc((size_t)NTOK * D_MODEL * 2);
  unsigned short* Win_bf  = (unsigned short*)wsalloc((size_t)D_IN_PROJ * D_MODEL * 2);
  unsigned short* Wout_bf = (unsigned short*)wsalloc((size_t)D_MODEL * D_INNER * 2);
  float*          xBC     = (float*)         wsalloc((size_t)NTOK * CONV_DIM * 4);
  float*          dt_s    = (float*)         wsalloc((size_t)NTOK * NHEADS * 4);
  float*          dA_s    = (float*)         wsalloc((size_t)NTOK * NHEADS * 4);
  float*          ybuf    = (float*)         wsalloc((size_t)NTOK * D_INNER * 4);
  unsigned short* g_bf    = (unsigned short*)wsalloc((size_t)NTOK * D_INNER * 2);

  // 1) convert operands to bf16 for the matrix cores
  cvt_f32_bf16<<<(NTOK * D_MODEL / 4 + 255) / 256, 256, 0, stream>>>(u, u_bf, NTOK * D_MODEL / 4);
  cvt_f32_bf16<<<(D_IN_PROJ * D_MODEL / 4 + 255) / 256, 256, 0, stream>>>(W_in, Win_bf, D_IN_PROJ * D_MODEL / 4);
  cvt_f32_bf16<<<(D_MODEL * D_INNER / 4 + 255) / 256, 256, 0, stream>>>(W_out, Wout_bf, D_MODEL * D_INNER / 4);

  // 2) in_proj GEMM (WMMA bf16): zxbcdt[8192][3352]
  {
    dim3 grid(NTOK / 128, (D_IN_PROJ + 63) / 64);
    gemm_bf16<<<grid, 256, 0, stream>>>(u_bf, Win_bf, zxbcdt, NTOK, D_IN_PROJ, D_MODEL);
  }

  // 3) depthwise conv + SiLU on xBC channels
  {
    int n = NTOK * CONV_DIM;
    conv_silu<<<(n + 255) / 256, 256, 0, stream>>>(zxbcdt, conv_w, conv_b, xBC);
  }

  // 4) dt / dA per (token, head)
  {
    int n = NTOK * NHEADS;
    dt_dA_kernel<<<(n + 255) / 256, 256, 0, stream>>>(zxbcdt, dt_bias, A_log, dt_s, dA_s);
  }

  // 5) selective scan (96 workgroups, full state in registers)
  scan_kernel<<<BATCH * NHEADS, 256, 0, stream>>>(xBC, dt_s, dA_s, Dparam, ybuf);

  // 6) RMSNorm + SiLU(z) gate -> bf16
  rmsnorm_gate<<<NTOK, 256, 0, stream>>>(ybuf, zxbcdt, norm_w, g_bf);

  // 7) out_proj GEMM (WMMA bf16) -> fp32 output
  {
    dim3 grid(NTOK / 128, (D_MODEL + 63) / 64);
    gemm_bf16<<<grid, 256, 0, stream>>>(g_bf, Wout_bf, out, NTOK, D_MODEL, D_INNER);
  }
}
